// Eyettention_pretrain_69664369541846
// MI455X (gfx1250) — matrix operand order repaired and checked by reading, not compiled, not run on hardware
//
#include <hip/hip_runtime.h>
#include <cstddef>
#include <cstdint>

// ---------------------------------------------------------------------------
// Eyettention pretrain forward for MI455X (gfx1250, wave32, WMMA bf16).
//
// bf16 storage + v_wmma_f32_16x16x32_bf16 f32-accumulate. Input projections
// of every LSTM hoisted out of the time loops as large parallel WMMA GEMMs;
// sequential recurrences run one wave per 16-row batch tile with h/c/gates in
// LDS. Weights stay resident in the 192MB L2 and are streamed as b128
// fragments with global_prefetch_b8. CDNA5 data movers:
//   - tensor_load_to_lds (TDM) stages encoder Whh (32KB) into LDS
//   - global_load_async_to_lds_b128 stages decoder h-state tiles into LDS
// ---------------------------------------------------------------------------

typedef __attribute__((ext_vector_type(16))) __bf16 v16bf;
typedef __attribute__((ext_vector_type(8)))  float  v8f;
typedef __attribute__((ext_vector_type(4)))  unsigned u32x4;
typedef __attribute__((ext_vector_type(8)))  unsigned u32x8;

#define BATCH 256
#define MAXSN 64
#define MAXSP 128
#define EMB   768
#define HID   128

__device__ inline float sigf(float x) { return 1.f / (1.f + __expf(-x)); }

// Generic LDS pointer -> 32-bit LDS byte offset (low 32 bits of the generic address).
__device__ inline unsigned lds_off(const void* p) {
  return (unsigned)(uintptr_t)p;
}

// Per-lane async global->LDS 16B copy (ASYNCcnt tracked).
__device__ inline void async_copy_b128(unsigned lds_addr, const void* gsrc) {
  unsigned long long ga = (unsigned long long)(uintptr_t)gsrc;
  asm volatile("global_load_async_to_lds_b128 %0, %1, off"
               :: "v"(lds_addr), "v"(ga) : "memory");
}
__device__ inline void wait_async0() {
  asm volatile("s_wait_asynccnt 0x0" ::: "memory");
}

// A fragment: 16x32 bf16 (M x K), row-major, leading dim ld.
// lane l (m=l&15): two contiguous 16B runs at K = (l>>4)*8 and 16+(l>>4)*8.
__device__ inline v16bf load_frag_a(const __bf16* p, int ld) {
  const int lane = threadIdx.x & 31;
  const int m = lane & 15;
  const int kh = (lane >> 4) << 3;
  union { uint4 q[2]; v16bf v; } u;
  u.q[0] = *(const uint4*)(p + (size_t)m * ld + kh);
  u.q[1] = *(const uint4*)(p + (size_t)m * ld + kh + 16);
  return u.v;
}

// B fragment: 32x16 (K x N) with weights stored W[N][K] row-major (B^T).
// lane l holds column n=l&15, two contiguous 16B runs at K = (l>>4)*8 and 16+(l>>4)*8.
__device__ inline v16bf load_frag_b(const __bf16* w, int ld) {
  const int lane = threadIdx.x & 31;
  const int n = lane & 15;
  const int ks = (lane >> 4) << 3;
  union { uint4 q[2]; v16bf v; } u;
  u.q[0] = *(const uint4*)(w + (size_t)n * ld + ks);
  u.q[1] = *(const uint4*)(w + (size_t)n * ld + ks + 16);
  return u.v;
}

__device__ inline v8f wmma_bf16(v16bf a, v16bf b, v8f c) {
  return __builtin_amdgcn_wmma_f32_16x16x32_bf16(false, a, false, b,
                                                 (short)0, c, false, false);
}

// Per-wave GEMM: C_lds[16][ldc] (cols [0,N)) = act(A1@W1^T (+A2@W2^T) + bias)
__device__ void wave_gemm_lds(const __bf16* A1, int lda1,
                              const __bf16* W1, int k1, int ldw1,
                              const __bf16* A2, int lda2,
                              const __bf16* W2, int k2, int ldw2,
                              const float* bias,
                              float* C, int ldc, int N, int relu) {
  const int lane = threadIdx.x & 31;
  const int n_lane = lane & 15;
  const int mbase = (lane >> 4) << 3;
  for (int nt = 0; nt < N; nt += 16) {
    v8f acc;
    float bv = bias ? bias[nt + n_lane] : 0.f;
#pragma unroll
    for (int r = 0; r < 8; ++r) acc[r] = bv;
    for (int k0 = 0; k0 < k1; k0 += 32) {
      __builtin_prefetch(W1 + (size_t)(nt + n_lane) * ldw1 + k0 + 512, 0, 1);
      v16bf a = load_frag_a(A1 + k0, lda1);
      v16bf b = load_frag_b(W1 + (size_t)nt * ldw1 + k0, ldw1);
      acc = wmma_bf16(a, b, acc);
    }
    if (A2) {
      for (int k0 = 0; k0 < k2; k0 += 32) {
        v16bf a = load_frag_a(A2 + k0, lda2);
        v16bf b = load_frag_b(W2 + (size_t)nt * ldw2 + k0, ldw2);
        acc = wmma_bf16(a, b, acc);
      }
    }
#pragma unroll
    for (int r = 0; r < 8; ++r) {
      float v = acc[r];
      if (relu) v = v > 0.f ? v : 0.f;
      C[(mbase + r) * ldc + nt + n_lane] = v;
    }
  }
}

// ------------------------- prep kernels ------------------------------------
__global__ void k_zero_b16(__bf16* p, long n) {
  long i = (long)blockIdx.x * 256 + threadIdx.x;
  if (i < n) p[i] = (__bf16)0.f;
}
__global__ void k_zero_f32(float* p, long n) {
  long i = (long)blockIdx.x * 256 + threadIdx.x;
  if (i < n) p[i] = 0.f;
}
// cast f32 weight [N][K] -> bf16 [Npad][Kpad], zero padded
__global__ void k_cast_pad(const float* src, __bf16* dst, int N, int K,
                           int Npad, int Kpad) {
  long i = (long)blockIdx.x * 256 + threadIdx.x;
  long tot = (long)Npad * Kpad;
  if (i >= tot) return;
  int n = (int)(i / Kpad), k = (int)(i % Kpad);
  dst[i] = (n < N && k < K) ? (__bf16)src[(size_t)n * K + k] : (__bf16)0.f;
}
__global__ void k_bias2(const float* a, const float* b, float* dst, int N,
                        int Npad) {
  int i = blockIdx.x * 256 + threadIdx.x;
  if (i >= Npad) return;
  dst[i] = (i < N) ? (a[i] + (b ? b[i] : 0.f)) : 0.f;
}

// Subword pooling via run-accumulation over sorted ids (no atomics).
__global__ void k_pool(const float* emb, const int* tok, const float* wemb,
                       const int* ids, __bf16* dst, int slen, int nwords) {
  int b = blockIdx.x;
  int e = blockIdx.y * 256 + threadIdx.x;
  float acc = 0.f;
  int cur = ids[(size_t)b * 128];
  for (int s = 0; s < slen; ++s) {
    int id = ids[(size_t)b * 128 + s];
    if (id != cur) {
      dst[((size_t)b * nwords + cur) * EMB + e] = (__bf16)acc;
      acc = 0.f;
      cur = id;
    }
    float v = emb ? emb[((size_t)b * 128 + s) * EMB + e]
                  : wemb[(size_t)tok[(size_t)b * 128 + s] * EMB + e];
    acc += v;
  }
  dst[((size_t)b * nwords + cur) * EMB + e] = (__bf16)acc;
}

__global__ void k_mask(const __bf16* pooled, float* mask) {
  int b = blockIdx.x, w = threadIdx.x;
  float s = 0.f;
  for (int e = 0; e < EMB; ++e) s += (float)pooled[((size_t)b * MAXSN + w) * EMB + e];
  mask[(size_t)b * MAXSN + w] = (s != 0.f) ? 1.f : 0.f;
}

// LayerNorm(sp_pooled + pos_emb[sp_pos]) in-place (bf16), one block per row
__global__ void k_ln(__bf16* spp, const int* sp_pos, const float* pos_emb,
                     const float* g, const float* bb) {
  __shared__ float red[128];
  int bt = blockIdx.x;
  int b = bt / 127, t = bt % 127;
  int pos = sp_pos[(size_t)b * 128 + t];
  float x[6];
#pragma unroll
  for (int i = 0; i < 6; ++i) {
    int e = threadIdx.x + i * 128;
    x[i] = (float)spp[(size_t)bt * EMB + e] + pos_emb[(size_t)pos * EMB + e];
  }
  float s = 0.f;
#pragma unroll
  for (int i = 0; i < 6; ++i) s += x[i];
  red[threadIdx.x] = s;
  __syncthreads();
  for (int st = 64; st > 0; st >>= 1) {
    if (threadIdx.x < st) red[threadIdx.x] += red[threadIdx.x + st];
    __syncthreads();
  }
  float mean = red[0] / (float)EMB;
  __syncthreads();
  float sq = 0.f;
#pragma unroll
  for (int i = 0; i < 6; ++i) { float d = x[i] - mean; sq += d * d; }
  red[threadIdx.x] = sq;
  __syncthreads();
  for (int st = 64; st > 0; st >>= 1) {
    if (threadIdx.x < st) red[threadIdx.x] += red[threadIdx.x + st];
    __syncthreads();
  }
  float inv = rsqrtf(red[0] / (float)EMB + 1e-12f);
#pragma unroll
  for (int i = 0; i < 6; ++i) {
    int e = threadIdx.x + i * 128;
    spp[(size_t)bt * EMB + e] = (__bf16)((x[i] - mean) * inv * g[e] + bb[e]);
  }
}

// ---------------------- big parallel x-projection GEMM ---------------------
__global__ void k_gemm_xproj(const __bf16* __restrict__ Ag,
                             const __bf16* __restrict__ W,
                             const float* __restrict__ bias,
                             float* __restrict__ C, int K, int N) {
  const int wave = threadIdx.x >> 5;
  const int lane = threadIdx.x & 31;
  const long r0 = ((long)blockIdx.x * 8 + wave) * 16;
  const int n_lane = lane & 15;
  const int mbase = (lane >> 4) << 3;
  for (int nt = 0; nt < N; nt += 16) {
    v8f acc;
    float bv = bias[nt + n_lane];
#pragma unroll
    for (int r = 0; r < 8; ++r) acc[r] = bv;
    for (int k0 = 0; k0 < K; k0 += 32) {
      __builtin_prefetch(W + (size_t)(nt + n_lane) * K + k0 + 512, 0, 1);
      v16bf a = load_frag_a(Ag + r0 * K + k0, K);
      v16bf b = load_frag_b(W + (size_t)nt * K + k0, K);
      acc = wmma_bf16(a, b, acc);
    }
#pragma unroll
    for (int r = 0; r < 8; ++r)
      C[(r0 + mbase + r) * (long)N + nt + n_lane] = acc[r];
  }
}

// ---------------------- encoder LSTM recurrence ----------------------------
// Whh (32KB) staged into LDS by the Tensor Data Mover: a 1-D D# descriptor
// (data_size=2B, tile_dim0 = tensor_dim0 = 16384) issued by wave 0, waited
// with s_wait_tensorcnt, then workgroup barrier.
#define ENC_SMEM (32768 + 8 * 22528)
__global__ void k_enc_recur(const float* __restrict__ gx,
                            const __bf16* __restrict__ Whh,
                            const __bf16* __restrict__ xres,
                            __bf16* __restrict__ xout, int dirofs, int rev) {
  extern __shared__ char smem[];
  __bf16* WhhS = (__bf16*)smem;  // [256][64] = 32KB at LDS offset of smem
  const int tid = threadIdx.x, wave = tid >> 5, lane = tid & 31;
  if (tid < 32) {  // one wave drives the TDM (EXEC ignored by tensor ops)
    unsigned long long ga = (unsigned long long)(uintptr_t)Whh;
    u32x4 g0;
    g0[0] = 0x1u;                                   // count=1, user D#
    g0[1] = lds_off(WhhS);                          // lds_addr
    g0[2] = (unsigned)(ga & 0xffffffffu);           // global_addr[31:0]
    g0[3] = ((unsigned)(ga >> 32) & 0x01ffffffu)    // global_addr[56:32]
            | 0x80000000u;                          // type=2 ("image")
    u32x8 g1;
    g1[0] = 0x00010000u;        // workgroup_mask=0, data_size=1 (2 bytes)
    g1[1] = 0x40000000u;        // tensor_dim0[15:0]=16384 in bits[31:16]
    g1[2] = 0x00010000u;        // tensor_dim1 = 1
    g1[3] = 0x40000000u;        // tile_dim0 = 16384 in bits[31:16]
    g1[4] = 0x00000001u;        // tile_dim1 = 1
    g1[5] = 16384u;             // tensor_dim0_stride[31:0]
    g1[6] = 0x40000000u;        // tensor_dim1_stride[15:0] in bits[31:16]
    g1[7] = 0u;
    asm volatile("tensor_load_to_lds %0, %1" :: "s"(g0), "s"(g1) : "memory");
    __builtin_amdgcn_s_wait_tensorcnt(0);
  }
  __syncthreads();
  char* wb = smem + 32768 + wave * 22528;
  __bf16* hS = (__bf16*)wb;            // [16][64]
  float* cS = (float*)(wb + 2048);     // [16][64]
  float* gS = (float*)(wb + 6144);     // [16][256]
  const int r0 = blockIdx.x * 128 + wave * 16;
  for (int i = lane; i < 1024; i += 32) { hS[i] = (__bf16)0.f; cS[i] = 0.f; }
  const int n_lane = lane & 15, mbase = (lane >> 4) << 3;
  for (int s = 0; s < 64; ++s) {
    int t = rev ? 63 - s : s;
    for (int nt = 0; nt < 256; nt += 16) {
      v8f acc;
#pragma unroll
      for (int r = 0; r < 8; ++r)
        acc[r] = gx[((size_t)(r0 + mbase + r) * MAXSN + t) * 256 + nt + n_lane];
#pragma unroll
      for (int k0 = 0; k0 < 64; k0 += 32) {
        v16bf a = load_frag_a(hS + k0, 64);
        v16bf b = load_frag_b(WhhS + nt * 64 + k0, 64);
        acc = wmma_bf16(a, b, acc);
      }
#pragma unroll
      for (int r = 0; r < 8; ++r) gS[(mbase + r) * 256 + nt + n_lane] = acc[r];
    }
    for (int i = lane; i < 1024; i += 32) {
      int m = i >> 6, col = i & 63;
      float gi = gS[m * 256 + col], gf = gS[m * 256 + 64 + col];
      float gg = gS[m * 256 + 128 + col], go = gS[m * 256 + 192 + col];
      float c = sigf(gf) * cS[i] + sigf(gi) * tanhf(gg);
      float h = sigf(go) * tanhf(c);
      cS[i] = c;
      hS[i] = (__bf16)h;
      size_t ob = ((size_t)(r0 + m) * MAXSN + t) * HID + dirofs + col;
      float res = xres ? (float)xres[ob] : 0.f;
      xout[ob] = (__bf16)(h + res);
    }
  }
}

// word_enc[b][w][0:128] = x, [128] = sn_word_len
__global__ void k_wenc(const __bf16* x, const float* wl, float* wenc) {
  size_t i = (size_t)blockIdx.x * 256 + threadIdx.x;
  if (i >= (size_t)BATCH * MAXSN * 129) return;
  size_t bw = i / 129;
  int d = (int)(i % 129);
  wenc[i] = (d < HID) ? (float)x[bw * HID + d] : wl[bw];
}

// ---------------------- persistent decoder ---------------------------------
struct DecArgs {
  const float* gx0;      // [B,127,512] cell0 x-proj (+ biases)
  const __bf16* whh;     // [8][512][128]
  const __bf16* wihR;    // [7][512][128] (cells 1..7)
  const float* bsum;     // [8][512]
  const __bf16* attnW;   const float* attnB;  // [144][128], [144]
  const __bf16* d1W;     const float* d1B;    // [512][288], [512]
  const __bf16* d2W;     const float* d2B;    // [256][512]
  const __bf16* d3W;     const float* d3B;    // [256][256]
  const __bf16* d4W;     const float* d4B;    // [256][256]
  const __bf16* d5W;     const float* d5B;    // [128][256]
  const float* wenc;     // [B,64,129]
  const float* mask;     // [B,64]
  const int* sp_pos;     // [B,128]
  __bf16* hst;           // [8][B][128] recurrent h, bf16 (async-staged)
  float* cst;            // [8][B][128]
  float* out;            // [B,127,125]
};

#define DEC_WAVE_SMEM 69632
#define DEC_SMEM (2 * DEC_WAVE_SMEM)
__global__ void k_dec(DecArgs A_) {
  extern __shared__ char smem[];
  const int wave = threadIdx.x >> 5, lane = threadIdx.x & 31;
  char* wb = smem + wave * DEC_WAVE_SMEM;
  __bf16* S_in = (__bf16*)wb;                 // [16][512] A staging
  __bf16* S_h = (__bf16*)(wb + 16384);        // [16][128] h state
  __bf16* S_cur = (__bf16*)(wb + 20480);      // [16][128] residual bf16
  float* S_curf = (float*)(wb + 24576);       // [16][128] residual f32
  float* S_g = (float*)(wb + 32768);          // [16][512] gates / GEMM out
  float* S_p = (float*)(wb + 65536);          // [16][64]  attn probs
  const int r0 = blockIdx.x * 32 + wave * 16;
  const int n_lane = lane & 15, mbase = (lane >> 4) << 3;
  const unsigned sh_base = lds_off(S_h);

  for (int t = 0; t < 127; ++t) {
    // ---------------- 8 stacked LSTM cells ----------------
    for (int l = 0; l < 8; ++l) {
      // async-stage h[l] tile (4KB bf16) global -> LDS, bypassing VGPRs
      {
        const __bf16* gsrc = A_.hst + ((size_t)l * BATCH + r0) * HID;
        for (int i = lane; i < 256; i += 32)
          async_copy_b128(sh_base + (unsigned)i * 16, gsrc + i * 8);
        wait_async0();
      }
      const __bf16* Whh = A_.whh + (size_t)l * 512 * 128;
      if (l == 0) {
        wave_gemm_lds(S_h, 128, Whh, 128, 128, nullptr, 0, nullptr, 0, 0,
                      nullptr, S_g, 512, 512, 0);
        for (int i = lane; i < 8192; i += 32) {
          int m = i >> 9, col = i & 511;
          S_g[i] += A_.gx0[((size_t)(r0 + m) * 127 + t) * 512 + col];
        }
      } else {
        const __bf16* Wih = A_.wihR + (size_t)(l - 1) * 512 * 128;
        wave_gemm_lds(S_h, 128, Whh, 128, 128, S_cur, 128, Wih, 128, 128,
                      A_.bsum + l * 512, S_g, 512, 512, 0);
      }
      for (int i = lane; i < 2048; i += 32) {
        int m = i >> 7, col = i & 127;
        size_t sb = ((size_t)l * BATCH + r0 + m) * HID + col;
        float gi = S_g[m * 512 + col], gf = S_g[m * 512 + 128 + col];
        float gg = S_g[m * 512 + 256 + col], go = S_g[m * 512 + 384 + col];
        float c = sigf(gf) * A_.cst[sb] + sigf(gi) * tanhf(gg);
        float h = sigf(go) * tanhf(c);
        A_.cst[sb] = c;
        A_.hst[sb] = (__bf16)h;
        float nv = (l < 2) ? h : (S_curf[i] + h);  // residual from cell 2 on
        S_curf[i] = nv;
        S_cur[i] = (__bf16)nv;
      }
    }
    // ---------------- local-window attention ----------------
    wave_gemm_lds(S_cur, 128, A_.attnW, 128, 128, nullptr, 0, nullptr, 0, 0,
                  A_.attnB, S_g, 512, 144, 0);
    {
      int m = lane & 15, wh = lane >> 4;
      int b = r0 + m;
      int pos = A_.sp_pos[(size_t)b * 128 + t];
      int left = pos - 1 < 0 ? 0 : pos - 1;
      int right = pos + 1 > 63 ? 63 : pos + 1;
      for (int w = wh * 32; w < wh * 32 + 32; ++w) {
        const float* we = A_.wenc + ((size_t)b * MAXSN + w) * 129;
        float s = 0.f;
        for (int d = 0; d < 129; ++d) s += S_g[m * 512 + d] * we[d];
        if (w < left || w > right || A_.mask[(size_t)b * MAXSN + w] == 0.f)
          s += -1e9f;
        S_p[m * 64 + w] = s;
      }
    }
    if (lane < 16) {
      int m = lane;
      float mx = -3e38f;
      for (int w = 0; w < 64; ++w) { float v = S_p[m * 64 + w]; mx = v > mx ? v : mx; }
      float sum = 0.f;
      for (int w = 0; w < 64; ++w) {
        float e = __expf(S_p[m * 64 + w] - mx);
        S_p[m * 64 + w] = e;
        sum += e;
      }
      float inv = 1.f / sum;
      for (int w = 0; w < 64; ++w) S_p[m * 64 + w] *= inv;
    }
    // ctx + concat(input8) + zero-pad  -> hc in S_in [16][288 of 512]
    for (int idx = lane; idx < 16 * 129; idx += 32) {
      int m = idx / 129, d = idx % 129;
      int b = r0 + m;
      float s = 0.f;
      for (int w = 0; w < 64; ++w)
        s += S_p[m * 64 + w] * A_.wenc[((size_t)b * MAXSN + w) * 129 + d];
      S_in[m * 512 + d] = (__bf16)s;
    }
    for (int i = lane; i < 2048; i += 32) {
      int m = i >> 7, col = i & 127;
      S_in[m * 512 + 129 + col] = (__bf16)S_curf[i];
    }
    for (int i = lane; i < 16 * 31; i += 32) {
      int m = i / 31, j = i % 31;
      S_in[m * 512 + 257 + j] = (__bf16)0.f;
    }
    // ---------------- MLP head ----------------
    wave_gemm_lds(S_in, 512, A_.d1W, 288, 288, nullptr, 0, nullptr, 0, 0,
                  A_.d1B, S_g, 512, 512, 1);
    for (int i = lane; i < 8192; i += 32) S_in[i] = (__bf16)S_g[i];
    wave_gemm_lds(S_in, 512, A_.d2W, 512, 512, nullptr, 0, nullptr, 0, 0,
                  A_.d2B, S_g, 512, 256, 1);
    for (int i = lane; i < 4096; i += 32) {
      int m = i >> 8, col = i & 255;
      S_in[m * 512 + col] = (__bf16)S_g[m * 512 + col];
    }
    wave_gemm_lds(S_in, 512, A_.d3W, 256, 256, nullptr, 0, nullptr, 0, 0,
                  A_.d3B, S_g, 512, 256, 1);
    for (int i = lane; i < 4096; i += 32) {
      int m = i >> 8, col = i & 255;
      S_in[m * 512 + col] = (__bf16)S_g[m * 512 + col];
    }
    wave_gemm_lds(S_in, 512, A_.d4W, 256, 256, nullptr, 0, nullptr, 0, 0,
                  A_.d4B, S_g, 512, 256, 1);
    for (int i = lane; i < 4096; i += 32) {
      int m = i >> 8, col = i & 255;
      S_in[m * 512 + col] = (__bf16)S_g[m * 512 + col];
    }
    // d5: [16][125] straight to global output
    for (int nt = 0; nt < 128; nt += 16) {
      v8f acc;
      float bv = A_.d5B[nt + n_lane];
#pragma unroll
      for (int r = 0; r < 8; ++r) acc[r] = bv;
      for (int k0 = 0; k0 < 256; k0 += 32) {
        v16bf a = load_frag_a(S_in + k0, 512);
        v16bf b = load_frag_b(A_.d5W + (size_t)nt * 256 + k0, 256);
        acc = wmma_bf16(a, b, acc);
      }
      int n = nt + n_lane;
      if (n < 125) {
#pragma unroll
        for (int r = 0; r < 8; ++r) {
          int b = r0 + mbase + r;
          A_.out[((size_t)b * 127 + t) * 125 + n] = acc[r];
        }
      }
    }
  }
}

// ---------------------------------------------------------------------------
extern "C" void kernel_launch(void* const* d_in, const int* in_sizes, int n_in,
                              void* d_out, int out_size, void* d_ws,
                              size_t ws_size, hipStream_t stream) {
  (void)in_sizes; (void)n_in; (void)out_size; (void)ws_size;
  // Input map: top-level dict insertion order; params pytree-flattened with
  // sorted dict keys (attn,d1..d5,dec,enc,ln_b,ln_g,pos_emb,word_emb; within
  // each cell dict sorted: Whh,Wih,bhh,bih; enc dirs sorted: b,f).
  const float* sn_bert_emb = (const float*)d_in[0];
  const float* sn_word_len = (const float*)d_in[1];
  const int pi = 2;
  const float* attnW_f = (const float*)d_in[pi + 0];
  const float* attnB_f = (const float*)d_in[pi + 1];
  const float* dW_f[5], *dB_f[5];
  for (int i = 0; i < 5; ++i) {
    dW_f[i] = (const float*)d_in[pi + 2 + 2 * i];
    dB_f[i] = (const float*)d_in[pi + 3 + 2 * i];
  }
  const int decbase = pi + 12;   // 14: per layer [Whh, Wih, bhh, bih]
  const int encbase = decbase + 32;  // 46: per layer [bWhh,bWih,bbhh,bbih,f...]
  const float* ln_b = (const float*)d_in[encbase + 64];
  const float* ln_g = (const float*)d_in[encbase + 65];
  const float* pos_emb = (const float*)d_in[encbase + 66];
  const float* word_emb = (const float*)d_in[encbase + 67];
  const int* word_ids_sn = (const int*)d_in[encbase + 68];
  const int* sp_emd = (const int*)d_in[encbase + 69];
  const int* sp_pos = (const int*)d_in[encbase + 70];
  const int* word_ids_sp = (const int*)d_in[encbase + 71];

  // ---- workspace carve ----
  char* Wp = (char*)d_ws;
  size_t off = 0;
  auto alloc = [&](size_t bytes) -> char* {
    char* p = Wp + off;
    off += (bytes + 255) & ~((size_t)255);
    return p;
  };
  __bf16* pooled = (__bf16*)alloc((size_t)BATCH * MAXSN * EMB * 2);
  __bf16* spp = (__bf16*)alloc((size_t)BATCH * 127 * EMB * 2);  // dec_in in-place
  float* maskw = (float*)alloc((size_t)BATCH * MAXSN * 4);
  __bf16* xA = (__bf16*)alloc((size_t)BATCH * MAXSN * HID * 2);
  __bf16* xB = (__bf16*)alloc((size_t)BATCH * MAXSN * HID * 2);
  float* gxf = (float*)alloc((size_t)BATCH * MAXSN * 256 * 4);
  float* gxb = (float*)alloc((size_t)BATCH * MAXSN * 256 * 4);
  float* wenc = (float*)alloc((size_t)BATCH * MAXSN * 129 * 4);
  float* gx0 = (float*)alloc((size_t)BATCH * 127 * 512 * 4);
  __bf16* hst = (__bf16*)alloc((size_t)8 * BATCH * HID * 2);
  float* cst = (float*)alloc((size_t)8 * BATCH * HID * 4);
  __bf16 *encWhhP[8][2], *encWihP[8][2];
  float* encBP[8][2];
  for (int l = 0; l < 8; ++l)
    for (int d = 0; d < 2; ++d) {
      encWhhP[l][d] = (__bf16*)alloc((size_t)256 * 64 * 2);
      encWihP[l][d] = (__bf16*)alloc((size_t)256 * (l ? 128 : 768) * 2);
      encBP[l][d] = (float*)alloc(256 * 4);
    }
  __bf16* decWhhP = (__bf16*)alloc((size_t)8 * 512 * 128 * 2);
  __bf16* decWih0P = (__bf16*)alloc((size_t)512 * 768 * 2);
  __bf16* decWihRP = (__bf16*)alloc((size_t)7 * 512 * 128 * 2);
  float* decBP = (float*)alloc((size_t)8 * 512 * 4);
  __bf16* attnWp = (__bf16*)alloc((size_t)144 * 128 * 2);
  float* attnBp = (float*)alloc(144 * 4);
  __bf16* d1Wp = (__bf16*)alloc((size_t)512 * 288 * 2);
  float* d1Bp = (float*)alloc(512 * 4);
  __bf16* d2Wp = (__bf16*)alloc((size_t)256 * 512 * 2);
  float* d2Bp = (float*)alloc(256 * 4);
  __bf16* d3Wp = (__bf16*)alloc((size_t)256 * 256 * 2);
  float* d3Bp = (float*)alloc(256 * 4);
  __bf16* d4Wp = (__bf16*)alloc((size_t)256 * 256 * 2);
  float* d4Bp = (float*)alloc(256 * 4);
  __bf16* d5Wp = (__bf16*)alloc((size_t)128 * 256 * 2);
  float* d5Bp = (float*)alloc(128 * 4);

  auto gz = [](long n) { return (unsigned)((n + 255) / 256); };
  // ---- zero state / pooled buffers (every call: deterministic) ----
  {
    long n1 = (long)BATCH * MAXSN * EMB;
    long n2 = (long)BATCH * 127 * EMB;
    k_zero_b16<<<gz(n1), 256, 0, stream>>>(pooled, n1);
    k_zero_b16<<<gz(n2), 256, 0, stream>>>(spp, n2);
    long ns = (long)8 * BATCH * HID;
    k_zero_b16<<<gz(ns), 256, 0, stream>>>(hst, ns);
    k_zero_f32<<<gz(ns), 256, 0, stream>>>(cst, ns);
  }
  // ---- weight cast/pad + bias combine ----
  auto castW = [&](const float* src, __bf16* dst, int N, int K, int Np, int Kp) {
    long tot = (long)Np * Kp;
    k_cast_pad<<<gz(tot), 256, 0, stream>>>(src, dst, N, K, Np, Kp);
  };
  for (int l = 0; l < 8; ++l)
    for (int d = 0; d < 2; ++d) {
      // flatten order within enc layer: b(Whh,Wih,bhh,bih), f(...). d:0=f,1=b
      int base = encbase + 8 * l + (d == 0 ? 4 : 0);
      int K = l ? 128 : 768;
      castW((const float*)d_in[base + 0], encWhhP[l][d], 256, 64, 256, 64);
      castW((const float*)d_in[base + 1], encWihP[l][d], 256, K, 256, K);
      k_bias2<<<1, 256, 0, stream>>>((const float*)d_in[base + 2],
                                     (const float*)d_in[base + 3],
                                     encBP[l][d], 256, 256);
    }
  for (int l = 0; l < 8; ++l) {
    int base = decbase + 4 * l;
    castW((const float*)d_in[base + 0], decWhhP + (size_t)l * 512 * 128,
          512, 128, 512, 128);
    if (l == 0)
      castW((const float*)d_in[base + 1], decWih0P, 512, 768, 512, 768);
    else
      castW((const float*)d_in[base + 1],
            decWihRP + (size_t)(l - 1) * 512 * 128, 512, 128, 512, 128);
    k_bias2<<<2, 256, 0, stream>>>((const float*)d_in[base + 2],
                                   (const float*)d_in[base + 3],
                                   decBP + (size_t)l * 512, 512, 512);
  }
  castW(attnW_f, attnWp, 129, 128, 144, 128);
  k_bias2<<<1, 256, 0, stream>>>(attnB_f, nullptr, attnBp, 129, 144);
  castW(dW_f[0], d1Wp, 512, 257, 512, 288);
  k_bias2<<<2, 256, 0, stream>>>(dB_f[0], nullptr, d1Bp, 512, 512);
  castW(dW_f[1], d2Wp, 256, 512, 256, 512);
  k_bias2<<<1, 256, 0, stream>>>(dB_f[1], nullptr, d2Bp, 256, 256);
  castW(dW_f[2], d3Wp, 256, 256, 256, 256);
  k_bias2<<<1, 256, 0, stream>>>(dB_f[2], nullptr, d3Bp, 256, 256);
  castW(dW_f[3], d4Wp, 256, 256, 256, 256);
  k_bias2<<<1, 256, 0, stream>>>(dB_f[3], nullptr, d4Bp, 256, 256);
  castW(dW_f[4], d5Wp, 125, 256, 128, 256);
  k_bias2<<<1, 256, 0, stream>>>(dB_f[4], nullptr, d5Bp, 125, 128);

  // ---- pooling / mask / decoder-input LN ----
  k_pool<<<dim3(BATCH, 3), 256, 0, stream>>>(sn_bert_emb, nullptr, nullptr,
                                             word_ids_sn, pooled, 128, MAXSN);
  k_pool<<<dim3(BATCH, 3), 256, 0, stream>>>(nullptr, sp_emd, word_emb,
                                             word_ids_sp, spp, 127, 127);
  k_mask<<<BATCH, 64, 0, stream>>>(pooled, maskw);
  k_ln<<<BATCH * 127, 128, 0, stream>>>(spp, sp_pos, pos_emb, ln_g, ln_b);

  // ---- encoder: 8 residual biLSTM layers ----
  __bf16 *xcur = xA, *xnxt = xB;
  for (int l = 0; l < 8; ++l) {
    const __bf16* Ain = (l == 0) ? pooled : xcur;
    int K = (l == 0) ? 768 : 128;
    k_gemm_xproj<<<(BATCH * MAXSN) / 128, 256, 0, stream>>>(
        Ain, encWihP[l][0], encBP[l][0], gxf, K, 256);
    k_gemm_xproj<<<(BATCH * MAXSN) / 128, 256, 0, stream>>>(
        Ain, encWihP[l][1], encBP[l][1], gxb, K, 256);
    const __bf16* res = (l >= 2) ? xcur : nullptr;
    k_enc_recur<<<2, 256, ENC_SMEM, stream>>>(gxf, encWhhP[l][0], res, xnxt, 0, 0);
    k_enc_recur<<<2, 256, ENC_SMEM, stream>>>(gxb, encWhhP[l][1], res, xnxt, 64, 1);
    __bf16* tmp = xcur; xcur = xnxt; xnxt = tmp;
  }
  k_wenc<<<gz((long)BATCH * MAXSN * 129), 256, 0, stream>>>(xcur, sn_word_len,
                                                            wenc);
  // ---- decoder cell-0 input projection (hoisted out of the scan) ----
  k_gemm_xproj<<<(BATCH * 127) / 128, 256, 0, stream>>>(spp, decWih0P, decBP,
                                                        gx0, 768, 512);
  // ---- persistent decoder scan ----
  DecArgs da;
  da.gx0 = gx0; da.whh = decWhhP; da.wihR = decWihRP; da.bsum = decBP;
  da.attnW = attnWp; da.attnB = attnBp;
  da.d1W = d1Wp; da.d1B = d1Bp; da.d2W = d2Wp; da.d2B = d2Bp;
  da.d3W = d3Wp; da.d3B = d3Bp; da.d4W = d4Wp; da.d4B = d4Bp;
  da.d5W = d5Wp; da.d5B = d5Bp;
  da.wenc = wenc; da.mask = maskw; da.sp_pos = sp_pos;
  da.hst = hst; da.cst = cst; da.out = (float*)d_out;
  k_dec<<<BATCH / 32, 64, DEC_SMEM, stream>>>(da);
}
